// MultiheadAttention_52905407152549
// MI455X (gfx1250) — matrix operand order repaired and checked by reading, not compiled, not run on hardware
//
#include <hip/hip_runtime.h>
#include <hip/hip_bf16.h>

#define HEADS 8
#define CPH 32
#define Bsz 2
#define Cch 256
#define Hh 48
#define Ww 48
#define Nn (Hh * Ww)          // 2304
#define EMB (HEADS * CPH)     // 256
#define BN (Bsz * Nn)         // 4608

typedef _Float16 half_t;
typedef _Float16 v8h  __attribute__((ext_vector_type(8)));
typedef _Float16 v16h __attribute__((ext_vector_type(16)));
typedef float    v8f  __attribute__((ext_vector_type(8)));
typedef int      gv4i __attribute__((vector_size(4 * sizeof(int))));
typedef int      gv2i __attribute__((vector_size(2 * sizeof(int))));

// ---------------------------------------------------------------------------
// CDNA5 async global->LDS path (ASYNCcnt), with safe fallback
// ---------------------------------------------------------------------------
#if defined(__gfx1250__) &&                                          \
    __has_builtin(__builtin_amdgcn_global_load_async_to_lds_b128) && \
    __has_builtin(__builtin_amdgcn_global_load_async_to_lds_b64) &&  \
    __has_builtin(__builtin_amdgcn_s_wait_asynccnt)
#define USE_ASYNC_LDS 1
#else
#define USE_ASYNC_LDS 0
#endif

__device__ __forceinline__ void async_copy_b128(const void* g, void* lds) {
#if USE_ASYNC_LDS
  __builtin_amdgcn_global_load_async_to_lds_b128(
      (__attribute__((address_space(1))) gv4i*)(uintptr_t)g,
      (__attribute__((address_space(3))) gv4i*)(uintptr_t)lds, 0, 0);
#else
  (void)g; (void)lds;
#endif
}

__device__ __forceinline__ void async_copy_b64(const void* g, void* lds) {
#if USE_ASYNC_LDS
  __builtin_amdgcn_global_load_async_to_lds_b64(
      (__attribute__((address_space(1))) gv2i*)(uintptr_t)g,
      (__attribute__((address_space(3))) gv2i*)(uintptr_t)lds, 0, 0);
#else
  (void)g; (void)lds;
#endif
}

__device__ __forceinline__ void wait_async_lds() {
#if USE_ASYNC_LDS
  __builtin_amdgcn_s_wait_asynccnt(0);
#endif
}

// ---------------------------------------------------------------------------
// WMMA helpers (CDNA5 wave32, 16x16x32 f16 -> f32)
// ---------------------------------------------------------------------------
__device__ __forceinline__ v8f wmma_f16(v16h a, v16h b, v8f c) {
  return __builtin_amdgcn_wmma_f32_16x16x32_f16(
      /*neg_a=*/false, a, /*neg_b=*/false, b,
      /*c_mod=*/(short)0, c, /*reuse_a=*/false, /*reuse_b=*/false);
}

// A-matrix fragment (16x32 f16, row-major source, ld halfs per row).
__device__ __forceinline__ v16h frag_a_load(const half_t* base, int ld, int lane) {
  const half_t* p = base + (size_t)(lane & 15) * ld + ((lane >> 4) * 8);
  v8h x0 = *(const v8h*)p;         // K = lo .. lo+7
  v8h x1 = *(const v8h*)(p + 16);  // K = 16+lo .. 16+lo+7
  v16h r;
#pragma unroll
  for (int i = 0; i < 8; ++i) { r[i] = x0[i]; r[i + 8] = x1[i]; }
  return r;
}

// B-matrix fragment (32x16 f16) gathered from a TRANSPOSED tile Bt[n][k].
__device__ __forceinline__ v16h frag_b_load_T(const half_t* baseT, int ld, int lane) {
  const half_t* p = baseT + (size_t)(lane & 15) * ld + ((lane >> 4) * 8);
  v8h x0 = *(const v8h*)p;
  v8h x1 = *(const v8h*)(p + 16);
  v16h r;
#pragma unroll
  for (int i = 0; i < 8; ++i) { r[i] = x0[i]; r[i + 8] = x1[i]; }
  return r;
}

__device__ __forceinline__ v8f v8f_zero() {
  v8f z;
#pragma unroll
  for (int i = 0; i < 8; ++i) z[i] = 0.0f;
  return z;
}

// ---------------------------------------------------------------------------
// Kernel 0: fp32 -> f16 weight conversion
// ---------------------------------------------------------------------------
__global__ __launch_bounds__(256) void mha_convert_weights(
    const float* __restrict__ wqkv, const float* __restrict__ wout,
    half_t* __restrict__ wqkv_h, half_t* __restrict__ wout_h) {
  int i = blockIdx.x * 256 + threadIdx.x;
  const int n1 = 3 * EMB * Cch;   // 196608
  const int n2 = Cch * EMB;       // 65536
  if (i < n1)            wqkv_h[i]      = (half_t)wqkv[i];
  else if (i < n1 + n2)  wout_h[i - n1] = (half_t)wout[i - n1];
}

// ---------------------------------------------------------------------------
// Kernel 1: GDN  y = x / sqrt(beta + gamma @ x^2), emit tokens t[B*N][C] f16
// ---------------------------------------------------------------------------
__global__ __launch_bounds__(256) void mha_gdn(
    const float* __restrict__ x, const float* __restrict__ beta,
    const float* __restrict__ gamma, half_t* __restrict__ t_out) {
  __shared__ float xs[16][Cch];
  __shared__ float x2s[16][Cch];
  const int blk = blockIdx.x;                 // 288 blocks
  const int b   = blk / (Nn / 16);
  const int n0  = (blk % (Nn / 16)) * 16;
  const int tid = threadIdx.x;
  const int lp  = tid & 15;
  const int cg  = tid >> 4;
  const float* xb = x + (size_t)b * Cch * Nn;
#pragma unroll
  for (int i = 0; i < 16; ++i) {
    int c = cg + 16 * i;
    float v = xb[(size_t)c * Nn + n0 + lp];
    xs[lp][c]  = v;
    x2s[lp][c] = v * v;
  }
  __syncthreads();
  const int d = tid;
  float acc[16];
#pragma unroll
  for (int p = 0; p < 16; ++p) acc[p] = 0.0f;
  const float* grow = gamma + (size_t)d * Cch;
  for (int c = 0; c < Cch; ++c) {
    float g = grow[c];
#pragma unroll
    for (int p = 0; p < 16; ++p) acc[p] += g * x2s[p][c];
  }
  const float bt = beta[d];
#pragma unroll
  for (int p = 0; p < 16; ++p) {
    float xn = xs[p][d] * rsqrtf(bt + acc[p]);
    t_out[(size_t)(b * Nn + n0 + p) * Cch + d] = (half_t)xn;
  }
}

// ---------------------------------------------------------------------------
// Kernel 2: QKV GEMM  [4608,256] x [256,768] -> q,k,v [B][h][N][32] f16
// ---------------------------------------------------------------------------
__global__ __launch_bounds__(256) void mha_qkv_gemm(
    const half_t* __restrict__ A, const half_t* __restrict__ Wq,
    half_t* __restrict__ qh, half_t* __restrict__ kh, half_t* __restrict__ vh) {
  __shared__ half_t As[128][32];
  __shared__ half_t Bt[64][32];
  const int m0 = blockIdx.x * 128;
  const int j0 = blockIdx.y * 64;
  const int tid = threadIdx.x, wave = tid >> 5, lane = tid & 31;
  v8f acc[4];
#pragma unroll
  for (int t = 0; t < 4; ++t) acc[t] = v8f_zero();

  const int ra = tid >> 1, sa = (tid & 1) * 16;   // A tile: 32B / thread
  const int jb = tid >> 2, sb = (tid & 3) * 8;    // B tile: 16B / thread

  for (int k0 = 0; k0 < Cch; k0 += 32) {
#if USE_ASYNC_LDS
    const half_t* srcA = A + (size_t)(m0 + ra) * Cch + k0 + sa;
    async_copy_b128(srcA, &As[ra][sa]);
    async_copy_b128(srcA + 8, &As[ra][sa + 8]);
    async_copy_b128(Wq + (size_t)(j0 + jb) * Cch + k0 + sb, &Bt[jb][sb]);
    wait_async_lds();
#else
    {
      const uint4* src = (const uint4*)(A + (size_t)(m0 + ra) * Cch + k0 + sa);
      uint4* dst = (uint4*)&As[ra][sa];
      dst[0] = src[0]; dst[1] = src[1];
    }
    *(uint4*)&Bt[jb][sb] = *(const uint4*)(Wq + (size_t)(j0 + jb) * Cch + k0 + sb);
#endif
    __syncthreads();
    v16h af = frag_a_load(&As[wave * 16][0], 32, lane);
#pragma unroll
    for (int ct = 0; ct < 4; ++ct) {
      v16h bf = frag_b_load_T(&Bt[ct * 16][0], 32, lane);
      acc[ct] = wmma_f16(af, bf, acc[ct]);
    }
    __syncthreads();
  }

  const int b = m0 / Nn;
#pragma unroll
  for (int ct = 0; ct < 4; ++ct) {
    int j = j0 + ct * 16 + (lane & 15);
    int which = j >> 8;
    int head = (j >> 5) & 7;
    int d = j & 31;
    half_t* dst = (which == 0) ? qh : ((which == 1) ? kh : vh);
    float mul = (which == 0) ? 0.17677669529663687f : 1.0f;
#pragma unroll
    for (int vv = 0; vv < 8; ++vv) {
      int m = m0 + wave * 16 + ((lane >> 4) * 8) + vv;
      int n = m - b * Nn;
      dst[((size_t)(b * HEADS + head) * Nn + n) * CPH + d] =
          (half_t)(acc[ct][vv] * mul);
    }
  }
}

// ---------------------------------------------------------------------------
// Kernel 3: streaming attention, online softmax + rel-pos bias gather.
// ---------------------------------------------------------------------------
__global__ __launch_bounds__(256) void mha_attention(
    const half_t* __restrict__ q, const half_t* __restrict__ k,
    const half_t* __restrict__ v, const float* __restrict__ table,
    half_t* __restrict__ res) {
  __shared__ half_t Ks[32][32];               // [key][d]
  __shared__ half_t Vst[32][32];              // transposed: [d][key]
  __shared__ half_t Plds[8][16][32];          // per-wave probs
  const int blk = blockIdx.x;                 // 288
  const int qb = blk % (Nn / 128);
  const int bh = blk / (Nn / 128);
  const int b = bh >> 3, h = bh & 7;
  const int tid = threadIdx.x, wave = tid >> 5, lane = tid & 31;
  const size_t head_off = (size_t)(b * HEADS + h) * Nn * CPH;
  const half_t* qbase = q + head_off;
  const half_t* kbase = k + head_off;
  const half_t* vbase = v + head_off;
  const int q0 = qb * 128 + wave * 16;

  v16h qf = frag_a_load(qbase + (size_t)q0 * CPH, CPH, lane);

  int qy[8], qx[8];
#pragma unroll
  for (int vv = 0; vv < 8; ++vv) {
    int qr = q0 + ((lane >> 4) * 8) + vv;
    qy[vv] = qr / Ww;  qx[vv] = qr - qy[vv] * Ww;
  }

  float rmax[8], rsum[8];
#pragma unroll
  for (int vv = 0; vv < 8; ++vv) { rmax[vv] = -3.0e38f; rsum[vv] = 0.0f; }
  v8f o0 = v8f_zero(), o1 = v8f_zero();

  const int r = tid >> 3;                     // key within chunk
  const int s = (tid & 7) * 4;                // d offset

  for (int m0 = 0; m0 < Nn; m0 += 32) {
    // K chunk: async global->LDS (row-major, consumed row-major)
#if USE_ASYNC_LDS
    async_copy_b64(kbase + (size_t)(m0 + r) * CPH + s, &Ks[r][s]);
#else
    *(uint2*)&Ks[r][s] = *(const uint2*)(kbase + (size_t)(m0 + r) * CPH + s);
#endif
    // V chunk: needs transpose on store -> explicit path
    {
      const half_t* vp = vbase + (size_t)(m0 + r) * CPH + s;
      Vst[s + 0][r] = vp[0]; Vst[s + 1][r] = vp[1];
      Vst[s + 2][r] = vp[2]; Vst[s + 3][r] = vp[3];
    }
    if (m0 + 32 < Nn) {
      __builtin_prefetch(kbase + (size_t)(m0 + 32 + r) * CPH + s, 0, 1);
      __builtin_prefetch(vbase + (size_t)(m0 + 32 + r) * CPH + s, 0, 1);
    }
    wait_async_lds();
    __syncthreads();

    v16h kf0 = frag_b_load_T(&Ks[0][0], 32, lane);
    v16h kf1 = frag_b_load_T(&Ks[16][0], 32, lane);
    v8f s0 = wmma_f16(qf, kf0, v8f_zero());
    v8f s1 = wmma_f16(qf, kf1, v8f_zero());

    int col0 = m0 + (lane & 15), col1 = col0 + 16;
    int my0 = col0 / Ww, mx0 = col0 - my0 * Ww;
    int my1 = col1 / Ww, mx1 = col1 - my1 * Ww;
#pragma unroll
    for (int vv = 0; vv < 8; ++vv) {
      int i0 = (qy[vv] - my0 + Hh - 1) * (2 * Ww - 1) + (qx[vv] - mx0 + Ww - 1);
      int i1 = (qy[vv] - my1 + Hh - 1) * (2 * Ww - 1) + (qx[vv] - mx1 + Ww - 1);
      s0[vv] += table[i0 * HEADS + h];
      s1[vv] += table[i1 * HEADS + h];
    }

#pragma unroll
    for (int vv = 0; vv < 8; ++vv) {
      float t = fmaxf(s0[vv], s1[vv]);
      t = fmaxf(t, __shfl_xor(t, 1));
      t = fmaxf(t, __shfl_xor(t, 2));
      t = fmaxf(t, __shfl_xor(t, 4));
      t = fmaxf(t, __shfl_xor(t, 8));
      float mnew = fmaxf(rmax[vv], t);
      float sc = __expf(rmax[vv] - mnew);
      rmax[vv] = mnew;
      float p0 = __expf(s0[vv] - mnew);
      float p1 = __expf(s1[vv] - mnew);
      s0[vv] = p0; s1[vv] = p1;
      float ps = p0 + p1;
      ps += __shfl_xor(ps, 1);
      ps += __shfl_xor(ps, 2);
      ps += __shfl_xor(ps, 4);
      ps += __shfl_xor(ps, 8);
      rsum[vv] = rsum[vv] * sc + ps;
      o0[vv] *= sc; o1[vv] *= sc;
    }

    {
      int rr = (lane >> 4) * 8, cc = lane & 15;
#pragma unroll
      for (int vv = 0; vv < 8; ++vv) {
        Plds[wave][rr + vv][cc]      = (half_t)s0[vv];
        Plds[wave][rr + vv][cc + 16] = (half_t)s1[vv];
      }
    }
    __syncthreads();

    v16h pf  = frag_a_load(&Plds[wave][0][0], 32, lane);
    v16h vf0 = frag_b_load_T(&Vst[0][0], 32, lane);
    v16h vf1 = frag_b_load_T(&Vst[16][0], 32, lane);
    o0 = wmma_f16(pf, vf0, o0);
    o1 = wmma_f16(pf, vf1, o1);
    __syncthreads();
  }

  {
    int rr = (lane >> 4) * 8, cc = lane & 15;
#pragma unroll
    for (int vv = 0; vv < 8; ++vv) {
      float inv = 1.0f / rsum[vv];
      size_t base = head_off + (size_t)(q0 + rr + vv) * CPH;
      res[base + cc]      = (half_t)(o0[vv] * inv);
      res[base + cc + 16] = (half_t)(o1[vv] * inv);
    }
  }
}

// ---------------------------------------------------------------------------
// Kernel 4: output GEMM  [4608,256] x [256,256] + bias, transposed store
// ---------------------------------------------------------------------------
__global__ __launch_bounds__(256) void mha_out_gemm(
    const half_t* __restrict__ A, const half_t* __restrict__ Wo,
    const float* __restrict__ b_out, float* __restrict__ out) {
  __shared__ half_t As[128][32];
  __shared__ half_t Bt[64][32];
  const int m0 = blockIdx.x * 128;
  const int j0 = blockIdx.y * 64;
  const int tid = threadIdx.x, wave = tid >> 5, lane = tid & 31;
  v8f acc[4];
#pragma unroll
  for (int t = 0; t < 4; ++t) acc[t] = v8f_zero();

  const int ra = tid >> 1, sa = (tid & 1) * 16;
  const int jb = tid >> 2, sb = (tid & 3) * 8;

  for (int k0 = 0; k0 < EMB; k0 += 32) {
#if USE_ASYNC_LDS
    const half_t* srcA = A + (size_t)(m0 + ra) * EMB + k0 + sa;
    async_copy_b128(srcA, &As[ra][sa]);
    async_copy_b128(srcA + 8, &As[ra][sa + 8]);
    async_copy_b128(Wo + (size_t)(j0 + jb) * EMB + k0 + sb, &Bt[jb][sb]);
    wait_async_lds();
#else
    {
      const uint4* src = (const uint4*)(A + (size_t)(m0 + ra) * EMB + k0 + sa);
      uint4* dst = (uint4*)&As[ra][sa];
      dst[0] = src[0]; dst[1] = src[1];
    }
    *(uint4*)&Bt[jb][sb] = *(const uint4*)(Wo + (size_t)(j0 + jb) * EMB + k0 + sb);
#endif
    __syncthreads();
    v16h af = frag_a_load(&As[wave * 16][0], 32, lane);
#pragma unroll
    for (int ct = 0; ct < 4; ++ct) {
      v16h bf = frag_b_load_T(&Bt[ct * 16][0], 32, lane);
      acc[ct] = wmma_f16(af, bf, acc[ct]);
    }
    __syncthreads();
  }

  const int b = m0 / Nn;
#pragma unroll
  for (int ct = 0; ct < 4; ++ct) {
    int c = j0 + ct * 16 + (lane & 15);
    float bo = b_out[c];
#pragma unroll
    for (int vv = 0; vv < 8; ++vv) {
      int m = m0 + wave * 16 + ((lane >> 4) * 8) + vv;
      int n = m - b * Nn;
      out[(size_t)b * Cch * Nn + (size_t)c * Nn + n] = acc[ct][vv] + bo;
    }
  }
}

// ---------------------------------------------------------------------------
// Launcher
// ---------------------------------------------------------------------------
extern "C" void kernel_launch(void* const* d_in, const int* in_sizes, int n_in,
                              void* d_out, int out_size, void* d_ws, size_t ws_size,
                              hipStream_t stream) {
  const float* x      = (const float*)d_in[0];
  const float* beta   = (const float*)d_in[1];
  const float* gamma  = (const float*)d_in[2];
  const float* w_qkv  = (const float*)d_in[3];
  const float* w_out  = (const float*)d_in[4];
  const float* b_out  = (const float*)d_in[5];
  const float* table  = (const float*)d_in[6];
  float* out = (float*)d_out;

  size_t off = 0;
  auto carve = [&](size_t bytes) -> void* {
    void* p = (char*)d_ws + off;
    off += (bytes + 255) & ~(size_t)255;
    return p;
  };
  half_t* wqkv_h = (half_t*)carve((size_t)3 * EMB * Cch * sizeof(half_t));
  half_t* wout_h = (half_t*)carve((size_t)Cch * EMB * sizeof(half_t));
  half_t* t_h    = (half_t*)carve((size_t)BN * Cch * sizeof(half_t));
  const size_t hsz = (size_t)Bsz * HEADS * Nn * CPH * sizeof(half_t);
  half_t* qh   = (half_t*)carve(hsz);
  half_t* kh   = (half_t*)carve(hsz);
  half_t* vh   = (half_t*)carve(hsz);
  half_t* resh = (half_t*)carve(hsz);

  mha_convert_weights<<<((3 * EMB * Cch + Cch * EMB) + 255) / 256, 256, 0, stream>>>(
      w_qkv, w_out, wqkv_h, wout_h);
  mha_gdn<<<Bsz * (Nn / 16), 256, 0, stream>>>(x, beta, gamma, t_h);
  mha_qkv_gemm<<<dim3(BN / 128, (3 * EMB) / 64), 256, 0, stream>>>(
      t_h, wqkv_h, qh, kh, vh);
  mha_attention<<<Bsz * HEADS * (Nn / 128), 256, 0, stream>>>(
      qh, kh, vh, table, resh);
  mha_out_gemm<<<dim3(BN / 128, Cch / 64), 256, 0, stream>>>(
      resh, wout_h, b_out, out);
}
